// GptOssAttention_3985729651607
// MI455X (gfx1250) — compile-verified, matrix operand order checked by reading, not run on hardware
//
#include <hip/hip_runtime.h>

typedef unsigned int u32;
typedef unsigned short bf16u;

typedef float  f32x4 __attribute__((ext_vector_type(4), may_alias));
typedef u32    u32x4 __attribute__((ext_vector_type(4), may_alias));
typedef u32    u32x2 __attribute__((ext_vector_type(2), may_alias));
typedef __bf16 bf16x16 __attribute__((ext_vector_type(16)));
typedef float  f32x8 __attribute__((ext_vector_type(8)));

#define S_LEN 4096
#define HID_DIM 2880
#define NH 64
#define NKV 8
#define DH 64
#define ATT_SCALE 0.125f

struct Pair128 { u32x4 lo, hi; };

__device__ __forceinline__ bf16u f2bf(float f) {
  u32 u = __float_as_uint(f);
  u32 r = (u + 0x7FFFu + ((u >> 16) & 1u)) >> 16;
  return (bf16u)r;
}

__device__ __forceinline__ bf16x16 make_frag(const bf16u* p0, const bf16u* p1) {
  Pair128 r;
  r.lo = *(const u32x4*)p0;
  r.hi = *(const u32x4*)p1;
  return __builtin_bit_cast(bf16x16, r);
}

__device__ __forceinline__ f32x8 wmma_bf16(bf16x16 a, bf16x16 b, f32x8 c) {
  return __builtin_amdgcn_wmma_f32_16x16x32_bf16(false, a, false, b, (short)0, c, false, false);
}

// ---------------- fp32 -> bf16 convert ----------------
__global__ void cvt4_kernel(const f32x4* __restrict__ src, u32x2* __restrict__ dst, long n4) {
  long i = (long)blockIdx.x * blockDim.x + threadIdx.x;
  if (i < n4) {
    f32x4 f = src[i];
    u32x2 o;
    o.x = (u32)f2bf(f.x) | ((u32)f2bf(f.y) << 16);
    o.y = (u32)f2bf(f.z) | ((u32)f2bf(f.w) << 16);
    dst[i] = o;
  }
}

// ---------------- bf16 GEMM: C[M][ldc] = A[M][K] * B[K][N] + bias ----------------
__global__ __launch_bounds__(256) void gemm_bf16_kernel(
    const bf16u* __restrict__ A, const bf16u* __restrict__ B,
    const float* __restrict__ bias, float* __restrict__ C,
    int M, int N, int K, int ldc)
{
  constexpr int LDS_S = 72;
  __shared__ __align__(16) bf16u As[128 * LDS_S];
  __shared__ __align__(16) bf16u Bs[128 * LDS_S];

  const int tid = threadIdx.x;
  const int wave = tid >> 5;
  const int lane = tid & 31;
  const int lr = lane & 15;
  const int grp = lane >> 4;
  const int wm = wave >> 1;   // 0..3
  const int wn = wave & 1;    // 0..1
  const int bm0 = blockIdx.y * 128;
  const int bn0 = blockIdx.x * 128;

  f32x8 acc[2][4];
#pragma unroll
  for (int i = 0; i < 2; i++)
#pragma unroll
    for (int j = 0; j < 4; j++) acc[i][j] = {};

  const int arow = tid >> 1;
  const int acb  = (tid & 1) * 32;
  const int bkr  = tid >> 2;
  const int bnb  = (tid & 3) * 32;

  const bf16u* aSrc = A + (long)(bm0 + arow) * K + acb;
  const bool bFast = (bn0 + bnb + 32) <= N;

  for (int k0 = 0; k0 < K; k0 += 64) {
    u32x4 areg[4];
    const u32x4* ap = (const u32x4*)(aSrc + k0);
#pragma unroll
    for (int j = 0; j < 4; j++) areg[j] = ap[j];

    u32x4 breg[4];
    bf16u bscal[32];
    const long brow = (long)(k0 + bkr) * N;
    if (bFast) {
      const u32x4* bp = (const u32x4*)(B + brow + bn0 + bnb);
#pragma unroll
      for (int j = 0; j < 4; j++) breg[j] = bp[j];
    } else {
#pragma unroll
      for (int e = 0; e < 32; e++) {
        int col = bn0 + bnb + e;
        bscal[e] = (col < N) ? B[brow + col] : (bf16u)0;
      }
    }
    if (k0 + 64 < K) {
      __builtin_prefetch(aSrc + k0 + 64, 0, 1);
      __builtin_prefetch(B + (long)(k0 + 64 + bkr) * N + bn0 + bnb, 0, 1);
    }

    __syncthreads();

    u32x4* ad = (u32x4*)(As + arow * LDS_S + acb);
#pragma unroll
    for (int j = 0; j < 4; j++) ad[j] = areg[j];

    if (bFast) {
#pragma unroll
      for (int j = 0; j < 4; j++) {
        u32 e0 = breg[j].x, e1 = breg[j].y, e2 = breg[j].z, e3 = breg[j].w;
        int nb0 = bnb + j * 8;
        Bs[(nb0 + 0) * LDS_S + bkr] = (bf16u)(e0 & 0xffff);
        Bs[(nb0 + 1) * LDS_S + bkr] = (bf16u)(e0 >> 16);
        Bs[(nb0 + 2) * LDS_S + bkr] = (bf16u)(e1 & 0xffff);
        Bs[(nb0 + 3) * LDS_S + bkr] = (bf16u)(e1 >> 16);
        Bs[(nb0 + 4) * LDS_S + bkr] = (bf16u)(e2 & 0xffff);
        Bs[(nb0 + 5) * LDS_S + bkr] = (bf16u)(e2 >> 16);
        Bs[(nb0 + 6) * LDS_S + bkr] = (bf16u)(e3 & 0xffff);
        Bs[(nb0 + 7) * LDS_S + bkr] = (bf16u)(e3 >> 16);
      }
    } else {
#pragma unroll
      for (int e = 0; e < 32; e++) Bs[(bnb + e) * LDS_S + bkr] = bscal[e];
    }

    __syncthreads();

#pragma unroll
    for (int kk = 0; kk < 64; kk += 32) {
      bf16x16 af[2], bfm[4];
#pragma unroll
      for (int ti = 0; ti < 2; ti++) {
        const bf16u* base = As + (wm * 32 + ti * 16 + lr) * LDS_S + kk + grp * 8;
        af[ti] = make_frag(base, base + 16);
      }
#pragma unroll
      for (int tj = 0; tj < 4; tj++) {
        const bf16u* base = Bs + (wn * 64 + tj * 16 + lr) * LDS_S + kk + grp * 8;
        bfm[tj] = make_frag(base, base + 16);
      }
#pragma unroll
      for (int ti = 0; ti < 2; ti++)
#pragma unroll
        for (int tj = 0; tj < 4; tj++)
          acc[ti][tj] = wmma_bf16(af[ti], bfm[tj], acc[ti][tj]);
    }
  }

#pragma unroll
  for (int ti = 0; ti < 2; ti++) {
#pragma unroll
    for (int tj = 0; tj < 4; tj++) {
      int col = bn0 + wn * 64 + tj * 16 + lr;
      if (col < N) {
        float bv = bias[col];
#pragma unroll
        for (int v = 0; v < 8; v++) {
          int row = bm0 + wm * 32 + ti * 16 + v + 8 * grp;
          C[(long)row * ldc + col] = acc[ti][tj][v] + bv;
        }
      }
    }
  }
}

// ---------------- RoPE + layout transform to bf16 [head][S][D] ----------------
__global__ void rope_qkv_kernel(const float* __restrict__ qkv, const float* __restrict__ fc,
                                bf16u* __restrict__ qo, bf16u* __restrict__ ko,
                                bf16u* __restrict__ vo)
{
  const int s = blockIdx.x;
  const int hh = blockIdx.y;   // 0..63 q, 64..71 k, 72..79 v
  const int i = threadIdx.x;   // 0..31
  const float* row = qkv + (long)s * 5120;
  if (hh < 64) {
    float cc = fc[s * 64 + i], sn = fc[s * 64 + 32 + i];
    float t1 = row[hh * 64 + i], t2 = row[hh * 64 + 32 + i];
    long o = ((long)hh * S_LEN + s) * DH;
    qo[o + i]      = f2bf(t1 * cc - t2 * sn);
    qo[o + 32 + i] = f2bf(t2 * cc + t1 * sn);
  } else if (hh < 72) {
    int kh = hh - 64;
    float cc = fc[s * 64 + i], sn = fc[s * 64 + 32 + i];
    float t1 = row[4096 + kh * 64 + i], t2 = row[4096 + kh * 64 + 32 + i];
    long o = ((long)kh * S_LEN + s) * DH;
    ko[o + i]      = f2bf(t1 * cc - t2 * sn);
    ko[o + 32 + i] = f2bf(t2 * cc + t1 * sn);
  } else {
    int kh = hh - 72;
    long o = ((long)kh * S_LEN + s) * DH;
    vo[o + i]      = f2bf(row[4608 + kh * 64 + i]);
    vo[o + 32 + i] = f2bf(row[4608 + kh * 64 + 32 + i]);
  }
}

// ---------------- Sliding-window flash attention with sinks ----------------
__global__ __launch_bounds__(256) void attn_kernel(
    const bf16u* __restrict__ qtab, const bf16u* __restrict__ ktab,
    const bf16u* __restrict__ vtab, const float* __restrict__ sinks,
    bf16u* __restrict__ out)
{
  constexpr int KSTR = 72;
  constexpr int VSTR = 264;
  constexpr int PSTR = 40;
  __shared__ __align__(16) bf16u Ks[256 * KSTR];
  __shared__ __align__(16) bf16u Vt[64 * VSTR];
  __shared__ __align__(16) bf16u Ps[8 * 16 * PSTR];

  const int nblk = blockIdx.x;   // 0..31
  const int h = blockIdx.y;      // 0..63
  const int kh = h >> 3;
  const int tid = threadIdx.x;
  const int wave = tid >> 5;
  const int lane = tid & 31;
  const int lr = lane & 15;
  const int grp = lane >> 4;

  // Stage K window (256 rows) and V^T into LDS
  {
    const int key = tid;
    int kpos = nblk * 128 - 128 + key;
    int srow = kpos < 0 ? 0 : kpos;  // masked later
    const u32x4* kp = (const u32x4*)(ktab + ((long)kh * S_LEN + srow) * DH);
    const u32x4* vp = (const u32x4*)(vtab + ((long)kh * S_LEN + srow) * DH);
    u32x4* kd = (u32x4*)(Ks + key * KSTR);
#pragma unroll
    for (int j = 0; j < 8; j++) kd[j] = kp[j];
#pragma unroll
    for (int j = 0; j < 8; j++) {
      u32x4 w = vp[j];
      int d0 = j * 8;
      Vt[(d0 + 0) * VSTR + key] = (bf16u)(w.x & 0xffff);
      Vt[(d0 + 1) * VSTR + key] = (bf16u)(w.x >> 16);
      Vt[(d0 + 2) * VSTR + key] = (bf16u)(w.y & 0xffff);
      Vt[(d0 + 3) * VSTR + key] = (bf16u)(w.y >> 16);
      Vt[(d0 + 4) * VSTR + key] = (bf16u)(w.z & 0xffff);
      Vt[(d0 + 5) * VSTR + key] = (bf16u)(w.z >> 16);
      Vt[(d0 + 6) * VSTR + key] = (bf16u)(w.w & 0xffff);
      Vt[(d0 + 7) * VSTR + key] = (bf16u)(w.w >> 16);
    }
  }

  // Q fragments (16 rows per wave), kept in registers
  bf16x16 aq[2];
  {
    int qrow = nblk * 128 + wave * 16 + lr;
    const bf16u* qp = qtab + ((long)h * S_LEN + qrow) * DH;
#pragma unroll
    for (int kk = 0; kk < 2; kk++)
      aq[kk] = make_frag(qp + kk * 32 + grp * 8, qp + kk * 32 + 16 + grp * 8);
  }

  __syncthreads();

  float m_run[8], l_run[8];
#pragma unroll
  for (int v = 0; v < 8; v++) { m_run[v] = -1e30f; l_run[v] = 0.f; }
  f32x8 oacc[4];
#pragma unroll
  for (int t = 0; t < 4; t++) oacc[t] = {};

  bf16u* psw = Ps + wave * 16 * PSTR;

  for (int c = 0; c < 8; c++) {           // 8 chunks of 32 keys = 256 window keys
    f32x8 s0 = {}, s1 = {};
#pragma unroll
    for (int kk = 0; kk < 2; kk++) {
      const bf16u* kr0 = Ks + (c * 32 + lr) * KSTR + kk * 32 + grp * 8;
      const bf16u* kr1 = Ks + (c * 32 + 16 + lr) * KSTR + kk * 32 + grp * 8;
      bf16x16 b0 = make_frag(kr0, kr0 + 16);
      bf16x16 b1 = make_frag(kr1, kr1 + 16);
      s0 = wmma_bf16(aq[kk], b0, s0);
      s1 = wmma_bf16(aq[kk], b1, s1);
    }

    float val0[8], val1[8];
#pragma unroll
    for (int v = 0; v < 8; v++) {
      int r = wave * 16 + v + 8 * grp;
      int c0 = c * 32 + lr;
      int c1 = c0 + 16;
      bool ok0 = (c0 > r) && (c0 <= r + 128) && ((nblk > 0) || (c0 >= 128));
      bool ok1 = (c1 > r) && (c1 <= r + 128) && ((nblk > 0) || (c1 >= 128));
      val0[v] = ok0 ? s0[v] * ATT_SCALE : -1e30f;
      val1[v] = ok1 ? s1[v] * ATT_SCALE : -1e30f;
    }

    float corr[8];
#pragma unroll
    for (int v = 0; v < 8; v++) {
      float m = fmaxf(val0[v], val1[v]);
#pragma unroll
      for (int d = 1; d < 16; d <<= 1) m = fmaxf(m, __shfl_xor(m, d, 32));
      float mnew = fmaxf(m_run[v], m);
      corr[v] = __expf(m_run[v] - mnew);
      float p0 = __expf(val0[v] - mnew);
      float p1 = __expf(val1[v] - mnew);
      psw[(v + 8 * grp) * PSTR + lr]      = f2bf(p0);
      psw[(v + 8 * grp) * PSTR + 16 + lr] = f2bf(p1);
      float ls = p0 + p1;
#pragma unroll
      for (int d = 1; d < 16; d <<= 1) ls += __shfl_xor(ls, d, 32);
      l_run[v] = l_run[v] * corr[v] + ls;
      m_run[v] = mnew;
    }

#pragma unroll
    for (int t = 0; t < 4; t++)
#pragma unroll
      for (int v = 0; v < 8; v++) oacc[t][v] = oacc[t][v] * corr[v];

    __syncthreads();  // fence: P stores -> P fragment loads

    bf16x16 pf = make_frag(psw + lr * PSTR + grp * 8, psw + lr * PSTR + 16 + grp * 8);
#pragma unroll
    for (int t = 0; t < 4; t++) {
      const bf16u* vr = Vt + (t * 16 + lr) * VSTR + c * 32 + grp * 8;
      bf16x16 bv = make_frag(vr, vr + 16);
      oacc[t] = wmma_bf16(pf, bv, oacc[t]);
    }
    __syncthreads();
  }

  // fold in sink, normalize, write bf16 output [S][H*D]
  float sk = sinks[h];
  float fac[8];
#pragma unroll
  for (int v = 0; v < 8; v++) {
    float mt = fmaxf(m_run[v], sk);
    float co = __expf(m_run[v] - mt);
    float lt = l_run[v] * co + __expf(sk - mt);
    fac[v] = co / lt;
  }
  const int qrow0 = nblk * 128 + wave * 16;
#pragma unroll
  for (int t = 0; t < 4; t++)
#pragma unroll
    for (int v = 0; v < 8; v++) {
      long row = qrow0 + v + 8 * grp;
      out[row * (NH * DH) + h * DH + t * 16 + lr] = f2bf(oacc[t][v] * fac[v]);
    }
}

// ---------------- host launcher ----------------
extern "C" void kernel_launch(void* const* d_in, const int* in_sizes, int n_in,
                              void* d_out, int out_size, void* d_ws, size_t ws_size,
                              hipStream_t stream) {
  const float* x     = (const float*)d_in[0];
  const float* fc    = (const float*)d_in[1];
  const float* qw    = (const float*)d_in[2];
  const float* qbias = (const float*)d_in[3];
  const float* kw    = (const float*)d_in[4];
  const float* kbias = (const float*)d_in[5];
  const float* vw    = (const float*)d_in[6];
  const float* vbias = (const float*)d_in[7];
  const float* ow    = (const float*)d_in[8];
  const float* obias = (const float*)d_in[9];
  const float* sinks = (const float*)d_in[10];
  float* outp = (float*)d_out;
  char* ws = (char*)d_ws;

  size_t off = 0;
  auto alloc = [&](size_t b) { size_t o = off; off += (b + 255) & ~(size_t)255; return o; };
  size_t o_xbf = alloc((size_t)S_LEN * HID_DIM * 2);
  size_t o_wq  = alloc((size_t)HID_DIM * 4096 * 2);
  size_t o_wk  = alloc((size_t)HID_DIM * 512 * 2);
  size_t o_wv  = alloc((size_t)HID_DIM * 512 * 2);
  size_t o_qkv = alloc((size_t)S_LEN * 5120 * 4);
  size_t o_qb  = alloc((size_t)NH * S_LEN * DH * 2);
  size_t o_kb  = alloc((size_t)NKV * S_LEN * DH * 2);
  size_t o_vb  = alloc((size_t)NKV * S_LEN * DH * 2);

  bf16u* xbf  = (bf16u*)(ws + o_xbf);
  bf16u* wqb  = (bf16u*)(ws + o_wq);
  bf16u* wkb  = (bf16u*)(ws + o_wk);
  bf16u* wvb  = (bf16u*)(ws + o_wv);
  float* qkv  = (float*)(ws + o_qkv);
  bf16u* qbb  = (bf16u*)(ws + o_qb);
  bf16u* kbb  = (bf16u*)(ws + o_kb);
  bf16u* vbb  = (bf16u*)(ws + o_vb);
  bf16u* attn = (bf16u*)(ws + 0);       // alias over xbf+wq (consumed before attention)
  bf16u* owb  = (bf16u*)(ws + o_qkv);   // alias over qkv (consumed by rope before this)

  auto cvt = [&](const float* s, bf16u* d, long n) {
    long n4 = n / 4;
    cvt4_kernel<<<dim3((unsigned)((n4 + 255) / 256)), dim3(256), 0, stream>>>(
        (const f32x4*)s, (u32x2*)d, n4);
  };

  cvt(x,  xbf, (long)S_LEN * HID_DIM);
  cvt(qw, wqb, (long)HID_DIM * 4096);
  cvt(kw, wkb, (long)HID_DIM * 512);
  cvt(vw, wvb, (long)HID_DIM * 512);

  gemm_bf16_kernel<<<dim3(32, 32), 256, 0, stream>>>(xbf, wqb, qbias, qkv,        S_LEN, 4096, HID_DIM, 5120);
  gemm_bf16_kernel<<<dim3(4, 32),  256, 0, stream>>>(xbf, wkb, kbias, qkv + 4096, S_LEN, 512,  HID_DIM, 5120);
  gemm_bf16_kernel<<<dim3(4, 32),  256, 0, stream>>>(xbf, wvb, vbias, qkv + 4608, S_LEN, 512,  HID_DIM, 5120);

  rope_qkv_kernel<<<dim3(S_LEN, 80), 32, 0, stream>>>(qkv, fc, qbb, kbb, vbb);

  attn_kernel<<<dim3(32, 64), 256, 0, stream>>>(qbb, kbb, vbb, sinks, attn);

  cvt(ow, owb, (long)4096 * HID_DIM);
  gemm_bf16_kernel<<<dim3((2880 + 127) / 128, 32), 256, 0, stream>>>(
      attn, owb, obias, outp, S_LEN, HID_DIM, 4096, HID_DIM);
}
